// CRF_78907139162441
// MI455X (gfx1250) — matrix operand order repaired and checked by reading, not compiled, not run on hardware
//
#include <hip/hip_runtime.h>

typedef __attribute__((ext_vector_type(16))) __bf16 bf16x16;
typedef __attribute__((ext_vector_type(8)))  __bf16 bf16x8;
typedef __attribute__((ext_vector_type(8)))  float  f32x8;

#define KTOT    65536
#define CHUNKS  32
#define KCHUNK  (KTOT / CHUNKS)    // 2048
#define KSTEP   64
#define NSTEPS  (KCHUNK / KSTEP)   // 32
#define LDSTR   88                 // bf16 row stride: 176B, 16B-aligned, conflict-free

static __device__ inline bf16x8 ld8(const __bf16* p) {
    return *(const bf16x8*)p;
}

static __device__ inline bf16x16 cat16(bf16x8 a, bf16x8 b) {
    return __builtin_shufflevector(a, b, 0,1,2,3,4,5,6,7,8,9,10,11,12,13,14,15);
}

// ---------------------------------------------------------------------------
// Kernel 1: batched Gram matrix G[b] = F[b] @ F[b]^T via bf16 hi/lo split WMMA.
// Grid: (CHUNKS, 8).  Block: 256 threads = 8 wave32.
// Each block handles one K-chunk of one batch; writes a 64x64 partial Gram.
// ---------------------------------------------------------------------------
__global__ __launch_bounds__(256)
void crf_gram(const float* __restrict__ A, float* __restrict__ partial) {
    __shared__ __bf16 sHi[64][LDSTR];
    __shared__ __bf16 sLo[64][LDSTR];

    const int bidx  = blockIdx.y;
    const int chunk = blockIdx.x;
    const int t     = threadIdx.x;

    const float* Fb = A + (size_t)bidx * 64 * KTOT;

    // -------- staging assignment: thread t loads row sr, 16 floats at cbase
    const int sr    = t >> 2;          // 0..63
    const int cbase = (t & 3) * 16;    // 0,16,32,48
    const float* srcBase = Fb + (size_t)sr * KTOT + (size_t)chunk * KCHUNK + cbase;

    // -------- compute assignment: wave w owns tiles (ma, nb0) and (ma, nb0+1)
    const int w    = t >> 5;           // 0..7
    const int lane = t & 31;
    const int h    = lane >> 4;        // half-wave
    const int lr   = lane & 15;        // row-in-tile / col-in-tile
    const int ma   = w & 3;
    const int nb0  = (w >> 2) << 1;

    const __bf16* aHiP  = &sHi[ma * 16 + lr][0];
    const __bf16* aLoP  = &sLo[ma * 16 + lr][0];
    const __bf16* b0HiP = &sHi[nb0 * 16 + lr][0];
    const __bf16* b0LoP = &sLo[nb0 * 16 + lr][0];
    const __bf16* b1HiP = &sHi[(nb0 + 1) * 16 + lr][0];
    const __bf16* b1LoP = &sLo[(nb0 + 1) * 16 + lr][0];

    f32x8 c0 = {0.f,0.f,0.f,0.f,0.f,0.f,0.f,0.f};
    f32x8 c1 = {0.f,0.f,0.f,0.f,0.f,0.f,0.f,0.f};

    for (int ks = 0; ks < NSTEPS; ++ks) {
        // ---- stage 64 rows x 64 K into LDS as hi/lo bf16 ----
        const float4* p = (const float4*)(srcBase + ks * KSTEP);
        float4 f0 = p[0], f1 = p[1], f2 = p[2], f3 = p[3];
        float xs[16] = {f0.x,f0.y,f0.z,f0.w, f1.x,f1.y,f1.z,f1.w,
                        f2.x,f2.y,f2.z,f2.w, f3.x,f3.y,f3.z,f3.w};
        bf16x8 hiA, hiB, loA, loB;
        #pragma unroll
        for (int i = 0; i < 8; ++i) {
            float x = xs[i];
            __bf16 hb = (__bf16)x;
            hiA[i] = hb;
            loA[i] = (__bf16)(x - (float)hb);
        }
        #pragma unroll
        for (int i = 0; i < 8; ++i) {
            float x = xs[8 + i];
            __bf16 hb = (__bf16)x;
            hiB[i] = hb;
            loB[i] = (__bf16)(x - (float)hb);
        }
        *(bf16x8*)&sHi[sr][cbase]     = hiA;
        *(bf16x8*)&sHi[sr][cbase + 8] = hiB;
        *(bf16x8*)&sLo[sr][cbase]     = loA;
        *(bf16x8*)&sLo[sr][cbase + 8] = loB;
        __syncthreads();

        // ---- two K=32 WMMA sub-steps ----
        #pragma unroll
        for (int s = 0; s < 2; ++s) {
            // A operand layout (16-bit A 16x32): half-wave h holds K = h*8+{0..7} and h*8+16+{0..7}
            const int ka = s * 32 + h * 8;
            bf16x16 aH = cat16(ld8(aHiP + ka), ld8(aHiP + ka + 16));
            bf16x16 aL = cat16(ld8(aLoP + ka), ld8(aLoP + ka + 16));
            // B operand layout (16-bit B 32x16): half-wave h holds K = h*8+{0..7} and h*8+16+{0..7}
            const int kb = s * 32 + h * 8;
            bf16x16 bH0 = cat16(ld8(b0HiP + kb), ld8(b0HiP + kb + 16));
            bf16x16 bL0 = cat16(ld8(b0LoP + kb), ld8(b0LoP + kb + 16));
            bf16x16 bH1 = cat16(ld8(b1HiP + kb), ld8(b1HiP + kb + 16));
            bf16x16 bL1 = cat16(ld8(b1LoP + kb), ld8(b1LoP + kb + 16));

            // dot = hi*hi + hi*lo + lo*hi   (lo*lo term ~2^-16 rel, dropped)
            c0 = __builtin_amdgcn_wmma_f32_16x16x32_bf16(false, aH, false, bH0, (short)0, c0, false, false);
            c0 = __builtin_amdgcn_wmma_f32_16x16x32_bf16(false, aH, false, bL0, (short)0, c0, false, false);
            c0 = __builtin_amdgcn_wmma_f32_16x16x32_bf16(false, aL, false, bH0, (short)0, c0, false, false);
            c1 = __builtin_amdgcn_wmma_f32_16x16x32_bf16(false, aH, false, bH1, (short)0, c1, false, false);
            c1 = __builtin_amdgcn_wmma_f32_16x16x32_bf16(false, aH, false, bL1, (short)0, c1, false, false);
            c1 = __builtin_amdgcn_wmma_f32_16x16x32_bf16(false, aL, false, bH1, (short)0, c1, false, false);
        }
        __syncthreads();
    }

    // ---- write partial Gram: C/D layout: VGPR v, half h -> M = v + 8h; N = lane&15
    float* outp = partial + (size_t)(bidx * CHUNKS + chunk) * 4096;
    #pragma unroll
    for (int v = 0; v < 8; ++v) {
        const int grow = ma * 16 + h * 8 + v;
        outp[grow * 64 + nb0 * 16 + lr]       = c0[v];
        outp[grow * 64 + (nb0 + 1) * 16 + lr] = c1[v];
    }
}

// ---------------------------------------------------------------------------
// Kernel 2: reduce partials -> G, build potentials, 10 mean-field iterations,
// emit out[b,j] = logits[b,j] + mean_i(e[i]).   Grid: 8 blocks, 256 threads.
// ---------------------------------------------------------------------------
__global__ __launch_bounds__(256)
void crf_finish(const float* __restrict__ partial,
                const float* __restrict__ logits,
                const float* __restrict__ W,
                float* __restrict__ out) {
    __shared__ float sPot[64][65];
    __shared__ float sN[64];
    __shared__ float sLg[64];
    __shared__ float sE[64];
    __shared__ float sMean;

    const int b = blockIdx.x;
    const int t = threadIdx.x;
    const float* pB = partial + (size_t)b * CHUNKS * 4096;

    // reduce 32 partial Gram matrices
    for (int idx = t; idx < 4096; idx += 256) {
        float s = 0.f;
        for (int c = 0; c < CHUNKS; ++c) s += pB[c * 4096 + idx];
        sPot[idx >> 6][idx & 63] = s;
    }
    __syncthreads();

    if (t < 64) {
        sN[t]  = sqrtf(sPot[t][t]);
        sLg[t] = logits[b * 64 + t];
        sE[t]  = 0.f;
    }
    __syncthreads();

    // pairwise potential = (G / (n_i*n_j + 1e-6)) * (W + W^T)/2
    for (int idx = t; idx < 4096; idx += 256) {
        const int i = idx >> 6, j = idx & 63;
        const float wsym = 0.5f * (W[i * 64 + j] + W[j * 64 + i]);
        sPot[i][j] = sPot[i][j] / (sN[i] * sN[j] + 1e-6f) * wsym;
    }
    __syncthreads();

    // l[b,x,y] = logits[y] + e[x];  probs[b,i,j] = sigmoid(l[b,j,i]) = sigmoid(logits[i]+e[j])
    // e_new[i] = sum_j (2*sigmoid(logits[i]+e[j]) - 1) * pot[i,j]
    for (int it = 0; it < 10; ++it) {
        float acc = 0.f;
        if (t < 64) {
            const float lg = sLg[t];
            #pragma unroll 4
            for (int j = 0; j < 64; ++j) {
                const float sgm = 1.0f / (1.0f + __expf(-(lg + sE[j])));
                acc += (2.0f * sgm - 1.0f) * sPot[t][j];
            }
        }
        __syncthreads();
        if (t < 64) sE[t] = acc;
        __syncthreads();
    }

    if (t == 0) {
        float s = 0.f;
        for (int i = 0; i < 64; ++i) s += sE[i];
        sMean = s * (1.0f / 64.0f);
    }
    __syncthreads();

    if (t < 64) out[b * 64 + t] = sLg[t] + sMean;  // mean over axis=1 of final l
}

extern "C" void kernel_launch(void* const* d_in, const int* in_sizes, int n_in,
                              void* d_out, int out_size, void* d_ws, size_t ws_size,
                              hipStream_t stream) {
    const float* a_inter = (const float*)d_in[0];   // [8, 64, 65536] fp32
    const float* logits  = (const float*)d_in[1];   // [8, 64]
    const float* W       = (const float*)d_in[2];   // [1, 64, 64]
    float* out = (float*)d_out;                     // [8, 64]
    float* partial = (float*)d_ws;                  // 8 * 32 * 4096 floats = 4 MB

    dim3 grid(CHUNKS, 8, 1);
    crf_gram<<<grid, 256, 0, stream>>>(a_inter, partial);
    crf_finish<<<8, 256, 0, stream>>>(partial, logits, W, out);
}
